// LSTMCell_q_49331994362494
// MI455X (gfx1250) — 8-bit matrix operand order repaired and checked by reading, compiled, not run on hardware
//
#include <hip/hip_runtime.h>
#include <hip/hip_bf16.h>
#include <math.h>
#include <stdint.h>

typedef int v8i __attribute__((ext_vector_type(8)));

#define BDIM 2048
#define KDIM 2048
#define HDIM 1024
#define NDIM 4096
#define STEPQ 0.03125f
#define INV_STEPQ 32.0f
#define LDA 80
#define LDB 80
#define KSTEPS (KDIM / 64)

__device__ __forceinline__ float fqf(float x) {
    return fminf(fmaxf(rintf(x * INV_STEPQ), -128.0f), 127.0f) * STEPQ;
}
__device__ __forceinline__ int q8(float x) {
    return (int)fminf(fmaxf(rintf(x * INV_STEPQ), -128.0f), 127.0f);
}
__device__ __forceinline__ float sigmoidf_(float x) {
    return 1.0f / (1.0f + __expf(-x));
}

__global__ void quant_xh_kernel(const float* __restrict__ X,
                                const float* __restrict__ h,
                                int* __restrict__ XhQ) {
    int idx = blockIdx.x * blockDim.x + threadIdx.x;
    int row = idx >> 9;
    int col4 = (idx & 511) << 2;
    float4 v = (col4 < 1024)
        ? *(const float4*)(X + (size_t)row * 1024 + col4)
        : *(const float4*)(h + (size_t)row * 1024 + (col4 - 1024));
    int p = (q8(v.x) & 255) | ((q8(v.y) & 255) << 8) |
            ((q8(v.z) & 255) << 16) | ((q8(v.w) & 255) << 24);
    XhQ[idx] = p;
}

__global__ void quant_flat_kernel(const float* __restrict__ src,
                                  int* __restrict__ dst) {
    int idx = blockIdx.x * blockDim.x + threadIdx.x;
    float4 v = *(const float4*)(src + (size_t)idx * 4);
    int p = (q8(v.x) & 255) | ((q8(v.y) & 255) << 8) |
            ((q8(v.z) & 255) << 16) | ((q8(v.w) & 255) << 24);
    dst[idx] = p;
}

__global__ void quant_bias_kernel(const float* __restrict__ b,
                                  float* __restrict__ bQ) {
    int idx = blockIdx.x * blockDim.x + threadIdx.x;
    bQ[idx] = fqf(b[idx]);
}

__device__ __forceinline__ void async_issue(unsigned int la, unsigned int lb,
                                            unsigned long long ga,
                                            unsigned long long gb) {
    asm volatile(
        "global_load_async_to_lds_b128 %0, %2, off\n\t"
        "global_load_async_to_lds_b128 %0, %2, off offset:16\n\t"
        "global_load_async_to_lds_b128 %1, %3, off"
        :: "v"(la), "v"(lb), "v"(ga), "v"(gb)
        : "memory");
}

__device__ __forceinline__ void kstep_compute(const char* aBuf, const char* bBuf,
                                              int waveRow, int laneLo, int laneHi,
                                              v8i acc[4]) {
    v8i Af;
    const char* ap = aBuf + waveRow * LDA + laneHi * 16;
    #pragma unroll
    for (int jj = 0; jj < 4; ++jj) {
        int2 w = *(const int2*)(ap + (jj & 1) * 8 + (jj >> 1) * 32);
        Af[2 * jj] = w.x;
        Af[2 * jj + 1] = w.y;
    }
    v8i Bf[4];
    const char* bp0 = bBuf + laneLo * LDB + laneHi * 16;
    #pragma unroll
    for (int t = 0; t < 4; ++t) {
        const char* bp = bp0 + t * (16 * LDB);
        int4 x0 = *(const int4*)bp;
        int4 x1 = *(const int4*)(bp + 32);
        Bf[t][0] = x0.x; Bf[t][1] = x0.y; Bf[t][2] = x0.z; Bf[t][3] = x0.w;
        Bf[t][4] = x1.x; Bf[t][5] = x1.y; Bf[t][6] = x1.z; Bf[t][7] = x1.w;
    }
    #pragma unroll
    for (int t = 0; t < 4; ++t) {
        acc[t] = __builtin_amdgcn_wmma_i32_16x16x64_iu8(
            true, Af, true, Bf[t], acc[t], false, false);
    }
}

__global__ __launch_bounds__(256)
void lstm_gemm_kernel(const char* __restrict__ XhQ,
                      const char* __restrict__ WQ,
                      const float* __restrict__ bQ,
                      const float* __restrict__ Cin,
                      float* __restrict__ Hout,
                      float* __restrict__ Cout) {
    __shared__ __align__(16) char aT[2][128 * LDA];
    __shared__ __align__(16) char bT[2][64 * LDB];

    const int tid = threadIdx.x;
    const int lane = tid & 31;
    const int wave = tid >> 5;
    const int laneLo = lane & 15;
    const int laneHi = lane >> 4;
    const int m0 = blockIdx.y * 128;
    const int j0 = blockIdx.x * 16;
    const int waveRow = wave * 16 + laneLo;

    v8i acc[4] = {};

    const int arow = tid >> 1;
    const int aoff = (tid & 1) * 32;
    const int brow = tid >> 2;
    const int boff = (tid & 3) * 16;

    unsigned long long gA =
        (unsigned long long)(uintptr_t)(XhQ + (size_t)(m0 + arow) * KDIM + aoff);
    unsigned long long gB =
        (unsigned long long)(uintptr_t)(WQ +
            (size_t)((brow >> 4) * 1024 + j0 + (brow & 15)) * KDIM + boff);

    const unsigned int ldsA0 = (unsigned int)(uintptr_t)&aT[0][arow * LDA + aoff];
    const unsigned int ldsA1 = (unsigned int)(uintptr_t)&aT[1][arow * LDA + aoff];
    const unsigned int ldsB0 = (unsigned int)(uintptr_t)&bT[0][brow * LDB + boff];
    const unsigned int ldsB1 = (unsigned int)(uintptr_t)&bT[1][brow * LDB + boff];

    async_issue(ldsA0, ldsB0, gA, gB);
    gA += 64;
    gB += 64;

    #pragma unroll 1
    for (int step = 0; step < KSTEPS; step += 2) {
        async_issue(ldsA1, ldsB1, gA, gB);
        gA += 64;
        gB += 64;
        asm volatile("s_wait_asynccnt 0x3" ::: "memory");
        __syncthreads();
        kstep_compute(&aT[0][0], &bT[0][0], waveRow, laneLo, laneHi, acc);
        __syncthreads();

        if (step + 2 < KSTEPS) {
            async_issue(ldsA0, ldsB0, gA, gB);
            gA += 64;
            gB += 64;
            asm volatile("s_wait_asynccnt 0x3" ::: "memory");
        } else {
            asm volatile("s_wait_asynccnt 0x0" ::: "memory");
        }
        __syncthreads();
        kstep_compute(&aT[1][0], &bT[1][0], waveRow, laneLo, laneHi, acc);
        __syncthreads();
    }

    const int j = j0 + laneLo;
    const float s2 = STEPQ * STEPQ;
    const float bi = bQ[j];
    const float bf = bQ[1024 + j];
    const float bg = bQ[2048 + j];
    const float bo = bQ[3072 + j];
    const int rowBase = m0 + wave * 16 + laneHi * 8;
    #pragma unroll
    for (int r = 0; r < 8; ++r) {
        const size_t off = (size_t)(rowBase + r) * HDIM + j;
        float ig = fqf(sigmoidf_((float)acc[0][r] * s2 + bi));
        float fg = fqf(sigmoidf_((float)acc[1][r] * s2 + bf));
        float gg = fqf(tanhf((float)acc[2][r] * s2 + bg));
        float og = fqf(sigmoidf_((float)acc[3][r] * s2 + bo));
        float cq = fqf(Cin[off]);
        float cn = cq * fg + ig * gg;
        float hn = fqf(tanhf(cn)) * og;
        Hout[off] = hn;
        Cout[off] = cn;
    }
}

extern "C" void kernel_launch(void* const* d_in, const int* in_sizes, int n_in,
                              void* d_out, int out_size, void* d_ws, size_t ws_size,
                              hipStream_t stream) {
    const float* X    = (const float*)d_in[0];
    const float* h    = (const float*)d_in[1];
    const float* c    = (const float*)d_in[2];
    const float* W    = (const float*)d_in[3];
    const float* bias = (const float*)d_in[4];

    char* ws   = (char*)d_ws;
    char* XhQ  = ws;
    char* WQ   = ws + (size_t)BDIM * KDIM;
    float* bQ  = (float*)(ws + (size_t)BDIM * KDIM + (size_t)NDIM * KDIM);

    float* Hout = (float*)d_out;
    float* Cout = Hout + (size_t)BDIM * HDIM;

    quant_xh_kernel<<<(BDIM * (KDIM / 4)) / 256, 256, 0, stream>>>(X, h, (int*)XhQ);
    quant_flat_kernel<<<(NDIM * (KDIM / 4)) / 256, 256, 0, stream>>>(W, (int*)WQ);
    quant_bias_kernel<<<NDIM / 256, 256, 0, stream>>>(bias, bQ);

    dim3 grid(HDIM / 16, BDIM / 128);
    lstm_gemm_kernel<<<grid, 256, 0, stream>>>(XhQ, WQ, bQ, c, Hout, Cout);
}
